// AttentionBlock_65652870086758
// MI455X (gfx1250) — matrix operand order repaired and checked by reading, not compiled, not run on hardware
//
#include <hip/hip_runtime.h>

#define B_ 8
#define C_ 512
#define N_ 1024
#define NH_ 8
#define HD_ 64
#define G_ 32

typedef _Float16 v16h __attribute__((ext_vector_type(16)));
typedef _Float16 v8h  __attribute__((ext_vector_type(8)));
typedef float    v8f  __attribute__((ext_vector_type(8)));
typedef unsigned int v4u __attribute__((ext_vector_type(4)));
typedef int      v8i  __attribute__((ext_vector_type(8)));
typedef int      v4i  __attribute__((ext_vector_type(4)));

__device__ __forceinline__ v16h ldfrag(const _Float16* p0, const _Float16* p1) {
  v8h lo = *(const v8h*)p0;
  v8h hi = *(const v8h*)p1;
  v16h r;
#pragma unroll
  for (int i = 0; i < 8; ++i) { r[i] = lo[i]; r[i + 8] = hi[i]; }
  return r;
}

__device__ __forceinline__ v8f wmma16(v16h a, v16h b, v8f c) {
  return __builtin_amdgcn_wmma_f32_16x16x32_f16(false, a, false, b, (short)0, c, false, false);
}

// LDS byte offset of a __shared__ pointer (low 32 bits of the generic address)
__device__ __forceinline__ unsigned lds_off(const void* p) {
  return (unsigned)(unsigned long long)p;
}

// CDNA5 async DMA: 16 bytes global -> LDS, tracked by ASYNCcnt
__device__ __forceinline__ void async_ld16(unsigned loff, const void* g) {
  asm volatile("global_load_async_to_lds_b128 %0, %1, off"
               :: "v"(loff), "v"((unsigned long long)(size_t)g) : "memory");
}
__device__ __forceinline__ void wait_async0() {
  asm volatile("s_wait_asynccnt 0x0" ::: "memory");
}

// CDNA5 TDM: load a 2-D tile (128 rows x 32 halves, row stride 512 halves)
// from global to LDS. D# built per ISA ch.8; tracked by TENSORcnt.
__device__ __forceinline__ void tdm_issue(unsigned long long ga, unsigned ldsoff) {
  v4u g0;
  g0[0] = 1u;                                  // count=1, user descriptor
  g0[1] = ldsoff;                              // lds_addr
  g0[2] = (unsigned)ga;                        // global_addr[31:0]
  g0[3] = (unsigned)(ga >> 32) | (2u << 30);   // global_addr[56:32] | type=2
  v8i g1;
  g1[0] = 1 << 16;                             // data_size = 2 bytes
  g1[1] = (int)(512u << 16);                   // tensor_dim0 = 512
  g1[2] = (int)(128u << 16);                   // tensor_dim1 = 128
  g1[3] = 32 << 16;                            // tile_dim0 = 32
  g1[4] = 128;                                 // tile_dim1 = 128
  g1[5] = 512;                                 // tensor_dim0_stride = 512
  g1[6] = 0;
  g1[7] = 0;
  v4i z4 = {0, 0, 0, 0};
  v8i z8 = {0, 0, 0, 0, 0, 0, 0, 0};
  __builtin_amdgcn_tensor_load_to_lds(g0, g1, z4, z4, z8, 0);
}
__device__ __forceinline__ void wait_tensor0() {
  __builtin_amdgcn_s_wait_tensorcnt(0);
}

// ---------------- kernel 1: weight f32 -> f16 ----------------
__global__ void k_cvt_w(const float* __restrict__ qkv_w, const float* __restrict__ proj_w,
                        _Float16* __restrict__ wq, _Float16* __restrict__ wp) {
  int i = blockIdx.x * 256 + threadIdx.x;
  if (i < 3 * C_ * C_) wq[i] = (_Float16)qkv_w[i];
  if (i < C_ * C_)     wp[i] = (_Float16)proj_w[i];
}

// ---------------- kernel 2: GroupNorm -> f16 [b][n][c] (transposed) ----------------
__global__ void k_groupnorm(const float* __restrict__ x, const float* __restrict__ gamma,
                            const float* __restrict__ beta, _Float16* __restrict__ h16) {
  int bg = blockIdx.x;
  int b = bg >> 5, g = bg & 31;
  const int CPG = C_ / G_;                      // 16
  const size_t base = ((size_t)b * C_ + (size_t)g * CPG) * N_;
  const int total = CPG * N_;                   // 16384
  float s = 0.f, s2 = 0.f;
  for (int idx = threadIdx.x; idx < total; idx += 256) {
    float v = x[base + idx];
    s += v; s2 += v * v;
  }
#pragma unroll
  for (int off = 1; off < 32; off <<= 1) {
    s  += __shfl_xor(s, off, 32);
    s2 += __shfl_xor(s2, off, 32);
  }
  __shared__ float rs[8], rs2[8];
  int lane = threadIdx.x & 31, wave = threadIdx.x >> 5;
  if (lane == 0) { rs[wave] = s; rs2[wave] = s2; }
  __syncthreads();
  __shared__ float smean, srstd;
  if (threadIdx.x == 0) {
    float S = 0.f, S2 = 0.f;
#pragma unroll
    for (int w = 0; w < 8; ++w) { S += rs[w]; S2 += rs2[w]; }
    float mean = S / (float)total;
    float var = S2 / (float)total - mean * mean;
    smean = mean;
    srstd = rsqrtf(var + 1e-5f);
  }
  __syncthreads();
  const float mean = smean, rstd = srstd;

  // pass 2: normalize + transpose [c][n] -> [b][n][c] via LDS tiles of 256 n
  __shared__ _Float16 sT[256][16];
  const int c = threadIdx.x >> 4;              // 0..15
  const int n0 = (threadIdx.x & 15) * 16;      // 0..240
  const float gm = gamma[g * CPG + c] * rstd;
  const float bt = beta[g * CPG + c];
  for (int nt = 0; nt < N_; nt += 256) {
#pragma unroll
    for (int i = 0; i < 16; ++i) {
      float v = (x[base + (size_t)c * N_ + nt + n0 + i] - mean) * gm + bt;
      sT[n0 + i][c] = (_Float16)v;
    }
    __syncthreads();
    v8h* dst = (v8h*)&h16[((size_t)b * N_ + nt + threadIdx.x) * C_ + g * CPG];
    dst[0] = *(v8h*)&sT[threadIdx.x][0];
    dst[1] = *(v8h*)&sT[threadIdx.x][8];
    __syncthreads();
  }
}

// ---------------- kernel 3: QKV GEMM (WMMA + TDM double-buffer) ----------------
// h: [b][n][c]; out[o,n] = sum_c W[o,c]*h[b,n,c] + bias[o]
// q,k -> [b][head][n][hd] (q scaled), v -> [b][C][n]
__global__ void __launch_bounds__(256, 1)
k_qkv_gemm(const _Float16* __restrict__ w16, const float* __restrict__ bias,
           const _Float16* __restrict__ h16,
           _Float16* __restrict__ q16, _Float16* __restrict__ k16,
           _Float16* __restrict__ v16) {
  const int b = blockIdx.z;
  const int mBase = blockIdx.y * 128;    // o
  const int nBase = blockIdx.x * 128;    // n
  __shared__ _Float16 sA[2][128 * 32];   // [m][k]
  __shared__ _Float16 sB[2][128 * 32];   // [n][k]
  const int t = threadIdx.x, lane = t & 31, wave = t >> 5;
  const int wm = wave & 3, wn = wave >> 2;

  const unsigned long long gaA = (unsigned long long)(size_t)(w16 + (size_t)mBase * C_);
  const unsigned long long gaB = (unsigned long long)(size_t)(h16 + ((size_t)b * N_ + nBase) * C_);
  const unsigned ldsA0 = lds_off(&sA[0][0]);
  const unsigned ldsB0 = lds_off(&sB[0][0]);

  v8f acc[2][4];
#pragma unroll
  for (int mi = 0; mi < 2; ++mi)
#pragma unroll
    for (int cs = 0; cs < 4; ++cs) acc[mi][cs] = (v8f){};

  const int k0a = (lane >> 4) * 8;
  const int k0b = (lane >> 4) * 8;

  if (wave == 0) { tdm_issue(gaA, ldsA0); tdm_issue(gaB, ldsB0); }
  for (int it = 0; it < 16; ++it) {
    if (wave == 0) wait_tensor0();
    __syncthreads();
    const int buf = it & 1;
    if (it + 1 < 16 && wave == 0) {
      unsigned boff = (unsigned)((buf ^ 1) * (128 * 32 * 2));
      tdm_issue(gaA + 64ull * (it + 1), ldsA0 + boff);
      tdm_issue(gaB + 64ull * (it + 1), ldsB0 + boff);
    }
    const _Float16* A = &sA[buf][0];
    const _Float16* Bm = &sB[buf][0];
    int m0 = (32 * wm + (lane & 15)) * 32;
    int m1 = (32 * wm + 16 + (lane & 15)) * 32;
    v16h a0 = ldfrag(&A[m0 + k0a], &A[m0 + k0a + 16]);
    v16h a1 = ldfrag(&A[m1 + k0a], &A[m1 + k0a + 16]);
#pragma unroll
    for (int cs = 0; cs < 4; ++cs) {
      int nc = (64 * wn + 16 * cs + (lane & 15)) * 32;
      v16h bf = ldfrag(&Bm[nc + k0b], &Bm[nc + k0b + 16]);
      acc[0][cs] = wmma16(a0, bf, acc[0][cs]);
      acc[1][cs] = wmma16(a1, bf, acc[1][cs]);
    }
  }

#pragma unroll
  for (int mi = 0; mi < 2; ++mi) {
    int obase = mBase + 32 * wm + 16 * mi + 8 * (lane >> 4);
    int tsel = obase >> 9, rem0 = obase & 511;
#pragma unroll
    for (int cs = 0; cs < 4; ++cs) {
      int n = nBase + 64 * wn + 16 * cs + (lane & 15);
      if (tsel == 2) {               // v: [b][C][n]
#pragma unroll
        for (int j = 0; j < 8; ++j) {
          float val = acc[mi][cs][j] + bias[obase + j];
          v16[((size_t)b * C_ + rem0 + j) * N_ + n] = (_Float16)val;
        }
      } else {                       // q/k: [b][head][n][hd], 8 contiguous channels
        _Float16* dst = (tsel == 0) ? q16 : k16;
        float sc = (tsel == 0) ? 0.125f : 1.0f;
        v8h pk;
#pragma unroll
        for (int j = 0; j < 8; ++j) pk[j] = (_Float16)((acc[mi][cs][j] + bias[obase + j]) * sc);
        int head = rem0 >> 6, hc0 = rem0 & 63;
        *(v8h*)&dst[(((size_t)b * NH_ + head) * N_ + n) * HD_ + hc0] = pk;
      }
    }
  }
}

// ---------------- kernel 4: flash attention (WMMA, async staged) ----------------
// q,k: [bh][n][hd]; v: [bh][hd][n]; o16: [b][n][C]
__global__ void __launch_bounds__(256, 1)
k_attention(const _Float16* __restrict__ q16, const _Float16* __restrict__ k16,
            const _Float16* __restrict__ v16, _Float16* __restrict__ o16) {
  const int head = blockIdx.y, bb = blockIdx.z;
  const int bh = bb * NH_ + head;
  const int i0 = blockIdx.x * 128;
  __shared__ _Float16 sQ[128 * 64];      // [i][c]
  __shared__ _Float16 sK[2][32 * 64];    // [j][c]
  __shared__ _Float16 sV[2][64 * 32];    // [c][j]
  __shared__ _Float16 sP[8 * 16 * 32];
  const int t = threadIdx.x, lane = t & 31, wave = t >> 5;
  const size_t qkbase = (size_t)bh * N_ * HD_;
  const size_t vbase  = (size_t)bh * HD_ * N_;

  auto stageKV = [&](int buf, int jj) {
    int idx = t * 8;
    async_ld16(lds_off(&sK[buf][idx]), k16 + qkbase + (size_t)jj * HD_ + idx);
    int c = idx >> 5, j0 = idx & 31;
    async_ld16(lds_off(&sV[buf][idx]), v16 + vbase + (size_t)c * N_ + jj + j0);
  };

  {
    int idx = t * 32;
    const _Float16* g = q16 + qkbase + (size_t)i0 * HD_ + idx;
    unsigned l = lds_off(&sQ[idx]);
    async_ld16(l, g); async_ld16(l + 16, g + 8);
    async_ld16(l + 32, g + 16); async_ld16(l + 48, g + 24);
  }
  stageKV(0, 0);
  wait_async0();
  __syncthreads();

  const int mrow = lane & 15;
  const int iRow = wave * 16 + mrow;
  const int k0a = (lane >> 4) * 16;
  const int k0b = (lane >> 4) * 16;
  v16h qa0 = ldfrag(&sQ[iRow * 64 + k0a],      &sQ[iRow * 64 + k0a + 8]);
  v16h qa1 = ldfrag(&sQ[iRow * 64 + 32 + k0a], &sQ[iRow * 64 + 32 + k0a + 8]);

  v8f oacc[4];
#pragma unroll
  for (int cs = 0; cs < 4; ++cs) oacc[cs] = (v8f){};
  float m_i[8], l_i[8];
#pragma unroll
  for (int r = 0; r < 8; ++r) { m_i[r] = -1e30f; l_i[r] = 0.f; }

  for (int it = 0; it < N_ / 32; ++it) {
    const int buf = it & 1;
    if (it + 1 < N_ / 32) stageKV(buf ^ 1, 32 * (it + 1));
    const _Float16* Kt = &sK[buf][0];
    const _Float16* Vt = &sV[buf][0];

    v8f s0 = {}, s1 = {};
    int jc = lane & 15;
    {
      v16h bk = ldfrag(&Kt[jc * 64 + k0b], &Kt[jc * 64 + k0b + 8]);
      s0 = wmma16(qa0, bk, s0);
      bk = ldfrag(&Kt[jc * 64 + 32 + k0b], &Kt[jc * 64 + 32 + k0b + 8]);
      s0 = wmma16(qa1, bk, s0);
      bk = ldfrag(&Kt[(jc + 16) * 64 + k0b], &Kt[(jc + 16) * 64 + k0b + 8]);
      s1 = wmma16(qa0, bk, s1);
      bk = ldfrag(&Kt[(jc + 16) * 64 + 32 + k0b], &Kt[(jc + 16) * 64 + 32 + k0b + 8]);
      s1 = wmma16(qa1, bk, s1);
    }

    float rmax[8], alpha[8], rsum[8];
#pragma unroll
    for (int r = 0; r < 8; ++r) rmax[r] = fmaxf(s0[r], s1[r]);
#pragma unroll
    for (int off = 1; off < 16; off <<= 1)
#pragma unroll
      for (int r = 0; r < 8; ++r) rmax[r] = fmaxf(rmax[r], __shfl_xor(rmax[r], off, 32));
#pragma unroll
    for (int r = 0; r < 8; ++r) {
      float mn = fmaxf(m_i[r], rmax[r]);
      alpha[r] = __expf(m_i[r] - mn);
      m_i[r] = mn;
      s0[r] = __expf(s0[r] - mn);
      s1[r] = __expf(s1[r] - mn);
      rsum[r] = s0[r] + s1[r];
    }
#pragma unroll
    for (int off = 1; off < 16; off <<= 1)
#pragma unroll
      for (int r = 0; r < 8; ++r) rsum[r] += __shfl_xor(rsum[r], off, 32);
#pragma unroll
    for (int r = 0; r < 8; ++r) l_i[r] = l_i[r] * alpha[r] + rsum[r];
#pragma unroll
    for (int cs = 0; cs < 4; ++cs)
#pragma unroll
      for (int r = 0; r < 8; ++r) oacc[cs][r] *= alpha[r];

    _Float16* pw = &sP[wave * 16 * 32];
#pragma unroll
    for (int r = 0; r < 8; ++r) {
      int row = r + ((lane >> 4) * 8);
      pw[row * 32 + (lane & 15)]      = (_Float16)s0[r];
      pw[row * 32 + 16 + (lane & 15)] = (_Float16)s1[r];
    }
    __builtin_amdgcn_wave_barrier();
    v16h pa = ldfrag(&pw[mrow * 32 + k0a], &pw[mrow * 32 + k0a + 8]);
#pragma unroll
    for (int cs = 0; cs < 4; ++cs) {
      int cc = cs * 16 + (lane & 15);
      v16h bv = ldfrag(&Vt[cc * 32 + k0b], &Vt[cc * 32 + k0b + 8]);
      oacc[cs] = wmma16(pa, bv, oacc[cs]);
    }
    wait_async0();
    __syncthreads();
  }

#pragma unroll
  for (int cs = 0; cs < 4; ++cs) {
    int c = head * HD_ + cs * 16 + (lane & 15);
#pragma unroll
    for (int r = 0; r < 8; ++r) {
      int i = i0 + wave * 16 + r + ((lane >> 4) * 8);
      o16[((size_t)bb * N_ + i) * C_ + c] = (_Float16)(oacc[cs][r] / l_i[r]);
    }
  }
}

// ---------------- kernel 5: proj GEMM + bias + residual (TDM staged) ----------------
// out[b,o,n] = x + sum_c Wp[o,c]*o16[b,n,c] + bias[o]
__global__ void __launch_bounds__(256, 1)
k_proj_gemm(const _Float16* __restrict__ w16, const float* __restrict__ bias,
            const _Float16* __restrict__ o16, const float* __restrict__ x,
            float* __restrict__ out) {
  const int b = blockIdx.z;
  const int mBase = blockIdx.y * 128;    // o
  const int nBase = blockIdx.x * 128;    // n
  __shared__ _Float16 sA[2][128 * 32];
  __shared__ _Float16 sB[2][128 * 32];
  const int t = threadIdx.x, lane = t & 31, wave = t >> 5;
  const int wm = wave & 3, wn = wave >> 2;

  const unsigned long long gaA = (unsigned long long)(size_t)(w16 + (size_t)mBase * C_);
  const unsigned long long gaB = (unsigned long long)(size_t)(o16 + ((size_t)b * N_ + nBase) * C_);
  const unsigned ldsA0 = lds_off(&sA[0][0]);
  const unsigned ldsB0 = lds_off(&sB[0][0]);

  v8f acc[2][4];
#pragma unroll
  for (int mi = 0; mi < 2; ++mi)
#pragma unroll
    for (int cs = 0; cs < 4; ++cs) acc[mi][cs] = (v8f){};

  const int k0a = (lane >> 4) * 8;
  const int k0b = (lane >> 4) * 8;

  if (wave == 0) { tdm_issue(gaA, ldsA0); tdm_issue(gaB, ldsB0); }
  for (int it = 0; it < 16; ++it) {
    if (wave == 0) wait_tensor0();
    __syncthreads();
    const int buf = it & 1;
    if (it + 1 < 16 && wave == 0) {
      unsigned boff = (unsigned)((buf ^ 1) * (128 * 32 * 2));
      tdm_issue(gaA + 64ull * (it + 1), ldsA0 + boff);
      tdm_issue(gaB + 64ull * (it + 1), ldsB0 + boff);
    }
    const _Float16* A = &sA[buf][0];
    const _Float16* Bm = &sB[buf][0];
    int m0 = (32 * wm + (lane & 15)) * 32;
    int m1 = (32 * wm + 16 + (lane & 15)) * 32;
    v16h a0 = ldfrag(&A[m0 + k0a], &A[m0 + k0a + 16]);
    v16h a1 = ldfrag(&A[m1 + k0a], &A[m1 + k0a + 16]);
#pragma unroll
    for (int cs = 0; cs < 4; ++cs) {
      int nc = (64 * wn + 16 * cs + (lane & 15)) * 32;
      v16h bf = ldfrag(&Bm[nc + k0b], &Bm[nc + k0b + 16]);
      acc[0][cs] = wmma16(a0, bf, acc[0][cs]);
      acc[1][cs] = wmma16(a1, bf, acc[1][cs]);
    }
  }

#pragma unroll
  for (int mi = 0; mi < 2; ++mi) {
#pragma unroll
    for (int cs = 0; cs < 4; ++cs) {
      int n = nBase + 64 * wn + 16 * cs + (lane & 15);
#pragma unroll
      for (int j = 0; j < 8; ++j) {
        int o = mBase + 32 * wm + 16 * mi + 8 * (lane >> 4) + j;
        size_t idx = ((size_t)b * C_ + o) * N_ + n;
        out[idx] = x[idx] + acc[mi][cs][j] + bias[o];
      }
    }
  }
}

extern "C" void kernel_launch(void* const* d_in, const int* in_sizes, int n_in,
                              void* d_out, int out_size, void* d_ws, size_t ws_size,
                              hipStream_t stream) {
  const float* x      = (const float*)d_in[0];
  const float* gamma  = (const float*)d_in[1];
  const float* beta   = (const float*)d_in[2];
  const float* qkv_w  = (const float*)d_in[3];
  const float* qkv_b  = (const float*)d_in[4];
  const float* proj_w = (const float*)d_in[5];
  const float* proj_b = (const float*)d_in[6];
  float* out = (float*)d_out;

  char* ws = (char*)d_ws;
  const size_t SZ = (size_t)B_ * C_ * N_ * sizeof(_Float16);   // 8 MB
  _Float16* h16  = (_Float16*)(ws);
  _Float16* q16  = (_Float16*)(ws + SZ);
  _Float16* k16  = (_Float16*)(ws + 2 * SZ);
  _Float16* v16  = (_Float16*)(ws + 3 * SZ);
  _Float16* o16  = (_Float16*)(ws + 4 * SZ);
  _Float16* wq16 = (_Float16*)(ws + 5 * SZ);
  _Float16* wp16 = (_Float16*)(ws + 5 * SZ + (size_t)3 * C_ * C_ * sizeof(_Float16));

  k_cvt_w<<<(3 * C_ * C_) / 256, 256, 0, stream>>>(qkv_w, proj_w, wq16, wp16);
  k_groupnorm<<<B_ * G_, 256, 0, stream>>>(x, gamma, beta, h16);
  k_qkv_gemm<<<dim3(N_ / 128, (3 * C_) / 128, B_), 256, 0, stream>>>(wq16, qkv_b, h16, q16, k16, v16);
  k_attention<<<dim3(N_ / 128, NH_, B_), 256, 0, stream>>>(q16, k16, v16, o16);
  k_proj_gemm<<<dim3(N_ / 128, C_ / 128, B_), 256, 0, stream>>>(wp16, proj_b, o16, x, out);
}
